// SGPEncoder_11158325035090
// MI455X (gfx1250) — matrix operand order repaired and checked by reading, not compiled, not run on hardware
//
#include <hip/hip_runtime.h>
#include <hip/hip_bf16.h>

typedef __attribute__((ext_vector_type(16))) _Float16     v16h;
typedef __attribute__((ext_vector_type(8)))  float        v8f;
typedef __attribute__((ext_vector_type(8)))  unsigned int v8u;

// ---------------------------------------------------------------------------
// WMMA helpers (CDNA5 gfx1250, wave32): D = A(16xK) * B(Kx16) + C, K=32 f16
// ---------------------------------------------------------------------------
__device__ __forceinline__ v8f wmma_f16(v16h a, v16h b, v8f c) {
  return __builtin_amdgcn_wmma_f32_16x16x32_f16(
      /*neg_a=*/false, a, /*neg_b=*/false, b,
      /*c_mod=*/(short)0, c, /*reuse_a=*/false, /*reuse_b=*/false);
}

// Hardware tanh (V_TANH_F32 is a gfx1250 TRANS op); libm fallback keeps compile safe.
__device__ __forceinline__ float fast_tanh(float x) {
#if __has_builtin(__builtin_amdgcn_tanhf)
  return __builtin_amdgcn_tanhf(x);
#else
  return tanhf(x);
#endif
}

// A-fragment (16x32, f16) gathered from a row-major tile `base` (ld = leading dim).
// ISA 7.12.2: m=lane&15; lanes 0-15 hold K={0..7,16..23}, lanes 16-31 hold
// K={8..15,24..31}; each VGPR packs a consecutive K pair (dword-aligned).
__device__ __forceinline__ v16h frag_a(const _Float16* base, int kbase, int ld) {
  const int lane  = threadIdx.x & 31;
  const int m     = lane & 15;
  const int khalf = (lane & 16) ? 8 : 0;
  v8u tmp;
#pragma unroll
  for (int v = 0; v < 8; ++v) {
    const int kk = (v < 4) ? (2 * v + khalf) : (16 + 2 * (v - 4) + khalf);
    tmp[v] = *(const unsigned int*)(base + m * ld + kbase + kk);
  }
  return __builtin_bit_cast(v16h, tmp);
}

// B-fragment (32x16, f16). B[k][n] = W[n][k] for row-major W (ld = row stride).
// Lanes 0-15 hold K={0..7,16..23}, lanes 16-31 hold K={8..15,24..31}; n = lane&15;
// each VGPR packs a consecutive K pair (same order as frag_a).
__device__ __forceinline__ v16h frag_b(const _Float16* base, int kbase, int ld) {
  const int lane = threadIdx.x & 31;
  const int n    = lane & 15;
  const int kofs = (lane & 16) ? 8 : 0;
  v8u tmp;
#pragma unroll
  for (int v = 0; v < 8; ++v) {
    const int k = kbase + ((v < 4) ? (2 * v + kofs) : (16 + 2 * (v - 4) + kofs));
    tmp[v] = *(const unsigned int*)(base + n * ld + k);
  }
  return __builtin_bit_cast(v16h, tmp);
}

union PackH2 {
  _Float16 h[2];
  unsigned int u;
};

// ---------------------------------------------------------------------------
// Small utility kernels
// ---------------------------------------------------------------------------
__global__ void zero_f32(float* p, int n) {
  int i = blockIdx.x * blockDim.x + threadIdx.x;
  if (i < n) p[i] = 0.f;
}

__global__ void cvt_to_f16(_Float16* dst, const float* __restrict__ src, int n) {
  int i = blockIdx.x * blockDim.x + threadIdx.x;
  if (i < n) dst[i] = (_Float16)src[i];
}

// ---------------------------------------------------------------------------
// Fused ESN: one block = 16 nodes, 16 waves (512 thr), wave w owns reservoir
// columns [16w,16w+16). W_res / W_in B-fragments live in registers for the
// whole 64-step scan. Per step: vectorized x stage, pre-gather ALL 10
// A-fragments, then issue 10 chained WMMAs back-to-back (no ds waits between).
// ---------------------------------------------------------------------------
__global__ __launch_bounds__(512) void esn_kernel(
    const float* __restrict__ x,            // [N,64,64]
    const _Float16* __restrict__ w_in_h,    // [256,64]
    const _Float16* __restrict__ w_res_h,   // [256,256]
    float* __restrict__ h_out) {            // [N,256]
  __shared__ _Float16 sh_h[16 * 256];  // current reservoir state tile (f16)
  __shared__ _Float16 sh_x[16 * 64];   // x[:,t,:] tile (f16)

  const int nodeBase = blockIdx.x * 16;
  const int wave     = threadIdx.x >> 5;
  const int lane     = threadIdx.x & 31;
  const int colBase  = wave * 16;

  for (int i = threadIdx.x; i < 16 * 256; i += 512) sh_h[i] = (_Float16)0.f;

  // Resident weight fragments for this wave's 16 output columns.
  v16h wres[8];
#pragma unroll
  for (int ks = 0; ks < 8; ++ks)
    wres[ks] = frag_b(w_res_h + colBase * 256, ks * 32, 256);
  v16h win[2];
#pragma unroll
  for (int ks = 0; ks < 2; ++ks)
    win[ks] = frag_b(w_in_h + colBase * 64, ks * 32, 64);

  // x staging addresses: thread -> one float2 (node row tn, float pair tj)
  const int tn = threadIdx.x >> 5;        // 0..15 node within tile
  const int tj = threadIdx.x & 31;        // 0..31 float-pair within row

  v8f hreg = {};  // running h (f32) for this wave's 16x16 tile
  __syncthreads();

  for (int t = 0; t < 64; ++t) {
    // stage x[nodeBase:+16, t, :] as f16 (global b64 load -> packed b32 LDS store)
    {
      const float2 xv = *(const float2*)(x + ((size_t)(nodeBase + tn) * 64 + t) * 64 + 2 * tj);
      PackH2 p;
      p.h[0] = (_Float16)xv.x;
      p.h[1] = (_Float16)xv.y;
      *(unsigned int*)(sh_x + tn * 64 + 2 * tj) = p.u;
    }
    __syncthreads();

    // pre-gather all A fragments, then chain the matrix pipe
    v16h afrag[10];
#pragma unroll
    for (int ks = 0; ks < 2; ++ks) afrag[ks] = frag_a(sh_x, ks * 32, 64);
#pragma unroll
    for (int ks = 0; ks < 8; ++ks) afrag[2 + ks] = frag_a(sh_h, ks * 32, 256);

    v8f acc = {};
#pragma unroll
    for (int ks = 0; ks < 2; ++ks) acc = wmma_f16(afrag[ks], win[ks], acc);
#pragma unroll
    for (int ks = 0; ks < 8; ++ks) acc = wmma_f16(afrag[2 + ks], wres[ks], acc);

#pragma unroll
    for (int v = 0; v < 8; ++v)      // leaky update: h = 0.7h + 0.3*tanh(.)
      hreg[v] = 0.7f * hreg[v] + 0.3f * fast_tanh(acc[v]);

    __syncthreads();                 // all reads of sh_h done
    {
      const int n = lane & 15, mofs = (lane & 16) ? 8 : 0;
#pragma unroll
      for (int v = 0; v < 8; ++v)
        sh_h[(v + mofs) * 256 + colBase + n] = (_Float16)hreg[v];
    }
    __syncthreads();
  }

  {
    const int n = lane & 15, mofs = (lane & 16) ? 8 : 0;
#pragma unroll
    for (int v = 0; v < 8; ++v)
      h_out[(size_t)(nodeBase + v + mofs) * 256 + colBase + n] = hreg[v];
  }
}

// ---------------------------------------------------------------------------
// Graph normalization + diffusion (sparse, edge-parallel)
// ---------------------------------------------------------------------------
__global__ void deg_kernel(const long long* __restrict__ ei, float* deg, int E) {
  int e = blockIdx.x * blockDim.x + threadIdx.x;
  if (e < E) {
    atomicAdd(&deg[(int)ei[e]], 1.f);
    atomicAdd(&deg[(int)ei[E + e]], 1.f);
  }
}

__global__ void dinv_kernel(const float* __restrict__ deg, float* dinv, int n) {
  int i = blockIdx.x * blockDim.x + threadIdx.x;
  if (i < n) {
    float d = deg[i];
    dinv[i] = d > 0.f ? fminf(rsqrtf(fmaxf(d, 1e-12f)), 1e6f) : 0.f;
  }
}

// one wave per directed edge: hout[r,:] += dinv[r]*dinv[c] * hin[c,:]
__global__ __launch_bounds__(256) void spmm_kernel(
    const long long* __restrict__ ei, const float* __restrict__ dinv,
    const float* __restrict__ hin, float* __restrict__ hout, int E) {
  const int w    = (int)((blockIdx.x * blockDim.x + threadIdx.x) >> 5);
  const int lane = threadIdx.x & 31;
  if (w >= 2 * E) return;
  int r, c;
  if (w < E) { r = (int)ei[w];           c = (int)ei[E + w]; }
  else       { r = (int)ei[E + (w - E)]; c = (int)ei[w - E]; }
  const float wt = dinv[r] * dinv[c];
  const float* src = hin + (size_t)c * 256;
  float*       dst = hout + (size_t)r * 256;
  __builtin_prefetch(src + lane, 0, 1);
#pragma unroll
  for (int j = 0; j < 8; ++j)
    atomicAdd(&dst[lane + j * 32], wt * src[lane + j * 32]);
}

// build f16 concat [h0|h1|h2|h3] : [N,1024]
__global__ void cat_cvt(_Float16* cat, const float* __restrict__ h0,
                        const float* __restrict__ h1, const float* __restrict__ h2,
                        const float* __restrict__ h3, int total) {
  int i = blockIdx.x * blockDim.x + threadIdx.x;
  if (i >= total) return;
  const int n = i >> 10, j = i & 1023;
  const float* h = (j < 256) ? h0 : (j < 512) ? h1 : (j < 768) ? h2 : h3;
  cat[i] = (_Float16)h[(size_t)n * 256 + (j & 255)];
}

// ---------------------------------------------------------------------------
// Fused readout: GEMM(1024->128) + exact GELU + GEMM(128->64) + LayerNorm.
// One block = 16 nodes, 8 waves. Waves own z1 column tiles; waves 0-3 own z2.
// k-loop processed in chunks of 4: gather 4 A + 4 B fragments, then 4 WMMAs.
// ---------------------------------------------------------------------------
__global__ __launch_bounds__(256) void readout_kernel(
    const _Float16* __restrict__ cat, const _Float16* __restrict__ w1_h,
    const float* __restrict__ b1, const _Float16* __restrict__ w2_h,
    const float* __restrict__ b2, const float* __restrict__ gamma,
    const float* __restrict__ beta, float* __restrict__ out) {
  __shared__ _Float16 sh_cat[16 * 1024];  // 32 KB
  __shared__ _Float16 sh_z1[16 * 128];
  __shared__ float    sh_z2[16 * 64];
  __shared__ float    sh_mu[16], sh_inv[16];

  const int nodeBase = blockIdx.x * 16;
  const int wave     = threadIdx.x >> 5;
  const int lane     = threadIdx.x & 31;

  {  // stage 16 concat rows (128-bit vectorized)
    const uint4* s = (const uint4*)(cat + (size_t)nodeBase * 1024);
    uint4*       d = (uint4*)sh_cat;
    for (int i = threadIdx.x; i < (16 * 1024) / 8; i += 256) d[i] = s[i];
  }
  __syncthreads();

  // z1 = GELU(cat @ W1^T + b1) : 128 cols, one 16-col tile per wave
  const int colBase = wave * 16;
  v8f acc = {};
#pragma unroll
  for (int kc = 0; kc < 8; ++kc) {   // 8 chunks x 4 k-steps = K=1024
    v16h af[4], bf[4];
#pragma unroll
    for (int u = 0; u < 4; ++u) {
      const int ks = kc * 4 + u;
      af[u] = frag_a(sh_cat, ks * 32, 1024);
      bf[u] = frag_b(w1_h + colBase * 1024, ks * 32, 1024);
    }
#pragma unroll
    for (int u = 0; u < 4; ++u) acc = wmma_f16(af[u], bf[u], acc);
  }
  {
    const int n = lane & 15, mofs = (lane & 16) ? 8 : 0;
#pragma unroll
    for (int v = 0; v < 8; ++v) {
      float val = acc[v] + b1[colBase + n];
      float g   = 0.5f * val * (1.f + erff(val * 0.70710678118f));
      sh_z1[(v + mofs) * 128 + colBase + n] = (_Float16)g;
    }
  }
  __syncthreads();

  // z2 = z1 @ W2^T + b2 : 64 cols -> waves 0..3
  if (wave < 4) {
    const int c2 = wave * 16;
    v16h af[4], bf[4];
#pragma unroll
    for (int ks = 0; ks < 4; ++ks) {
      af[ks] = frag_a(sh_z1, ks * 32, 128);
      bf[ks] = frag_b(w2_h + c2 * 128, ks * 32, 128);
    }
    v8f acc2 = {};
#pragma unroll
    for (int ks = 0; ks < 4; ++ks) acc2 = wmma_f16(af[ks], bf[ks], acc2);
    const int n = lane & 15, mofs = (lane & 16) ? 8 : 0;
#pragma unroll
    for (int v = 0; v < 8; ++v)
      sh_z2[(v + mofs) * 64 + c2 + n] = acc2[v] + b2[c2 + n];
  }
  __syncthreads();

  if (threadIdx.x < 16) {  // LayerNorm stats (biased var, eps=1e-5)
    float s = 0.f, ss = 0.f;
    for (int j = 0; j < 64; ++j) {
      float v = sh_z2[threadIdx.x * 64 + j];
      s += v; ss += v * v;
    }
    float mu  = s * (1.f / 64.f);
    float var = ss * (1.f / 64.f) - mu * mu;
    sh_mu[threadIdx.x]  = mu;
    sh_inv[threadIdx.x] = rsqrtf(var + 1e-5f);
  }
  __syncthreads();

  for (int i = threadIdx.x; i < 16 * 64; i += 256) {
    const int m = i >> 6, c = i & 63;
    out[(size_t)(nodeBase + m) * 64 + c] =
        (sh_z2[i] - sh_mu[m]) * sh_inv[m] * gamma[c] + beta[c];
  }
}

// ---------------------------------------------------------------------------
extern "C" void kernel_launch(void* const* d_in, const int* in_sizes, int n_in,
                              void* d_out, int out_size, void* d_ws, size_t ws_size,
                              hipStream_t stream) {
  const float*     x     = (const float*)d_in[0];
  const long long* ei    = (const long long*)d_in[1];  // int64 [2,E]
  const float*     W_in  = (const float*)d_in[2];
  const float*     W_res = (const float*)d_in[3];
  const float*     W1    = (const float*)d_in[4];
  const float*     b1    = (const float*)d_in[5];
  const float*     W2    = (const float*)d_in[6];
  const float*     b2    = (const float*)d_in[7];
  const float*     gamma = (const float*)d_in[8];
  const float*     beta  = (const float*)d_in[9];

  const int N = 8192, R = 256, E = 131072;

  char*  ws  = (char*)d_ws;
  size_t off = 0;
  auto alloc = [&](size_t bytes) -> void* {
    void* p = ws + off;
    off = (off + bytes + 255) & ~(size_t)255;
    return p;
  };
  _Float16* w_in_h  = (_Float16*)alloc((size_t)R * 64 * 2);
  _Float16* w_res_h = (_Float16*)alloc((size_t)R * R * 2);
  _Float16* w1_h    = (_Float16*)alloc((size_t)128 * 1024 * 2);
  _Float16* w2_h    = (_Float16*)alloc((size_t)64 * 128 * 2);
  float*    h0      = (float*)alloc((size_t)N * R * 4);
  float*    h1      = (float*)alloc((size_t)N * R * 4);
  float*    h2      = (float*)alloc((size_t)N * R * 4);
  float*    h3      = (float*)alloc((size_t)N * R * 4);
  float*    deg     = (float*)alloc((size_t)N * 4);
  float*    dinv    = (float*)alloc((size_t)N * 4);
  _Float16* cat     = (_Float16*)alloc((size_t)N * 1024 * 2);

  // 1) weights -> f16
  cvt_to_f16<<<(R * 64 + 255) / 256, 256, 0, stream>>>(w_in_h, W_in, R * 64);
  cvt_to_f16<<<(R * R + 255) / 256, 256, 0, stream>>>(w_res_h, W_res, R * R);
  cvt_to_f16<<<(128 * 1024 + 255) / 256, 256, 0, stream>>>(w1_h, W1, 128 * 1024);
  cvt_to_f16<<<(64 * 128 + 255) / 256, 256, 0, stream>>>(w2_h, W2, 64 * 128);

  // 2) fused ESN scan -> h0
  esn_kernel<<<N / 16, 512, 0, stream>>>(x, w_in_h, w_res_h, h0);

  // 3) degree + D^{-1/2}
  zero_f32<<<(N + 255) / 256, 256, 0, stream>>>(deg, N);
  deg_kernel<<<(E + 255) / 256, 256, 0, stream>>>(ei, deg, E);
  dinv_kernel<<<(N + 255) / 256, 256, 0, stream>>>(deg, dinv, N);

  // 4) three diffusion hops (sparse): h_{k+1} = A_hat @ h_k
  const int NR = N * R;
  const int spmmBlocks = (2 * E) / 8;  // 8 waves per 256-thread block
  zero_f32<<<(NR + 255) / 256, 256, 0, stream>>>(h1, NR);
  spmm_kernel<<<spmmBlocks, 256, 0, stream>>>(ei, dinv, h0, h1, E);
  zero_f32<<<(NR + 255) / 256, 256, 0, stream>>>(h2, NR);
  spmm_kernel<<<spmmBlocks, 256, 0, stream>>>(ei, dinv, h1, h2, E);
  zero_f32<<<(NR + 255) / 256, 256, 0, stream>>>(h3, NR);
  spmm_kernel<<<spmmBlocks, 256, 0, stream>>>(ei, dinv, h2, h3, E);

  // 5) concat -> f16
  cat_cvt<<<(N * 1024 + 255) / 256, 256, 0, stream>>>(cat, h0, h1, h2, h3, N * 1024);

  // 6) fused readout MLP + LayerNorm -> d_out [N,64] f32
  readout_kernel<<<N / 16, 256, 0, stream>>>(cat, w1_h, b1, w2_h, b2, gamma, beta,
                                             (float*)d_out);
}
